// TokenSwapMamba_56727928045683
// MI455X (gfx1250) — matrix operand order repaired and checked by reading, not compiled, not run on hardware
//
#include <hip/hip_runtime.h>
#include <hip/hip_bf16.h>
#include <math.h>

// ---------------------------------------------------------------------------
// Problem constants (match reference)
// ---------------------------------------------------------------------------
#define BATCH     2
#define SEQ       1024
#define DM        512          // d_model
#define DI        1024         // d_inner
#define DS        16           // d_state
#define DCONV     4
#define DTR       32           // dt_rank
#define XDBL      (DTR + 2*DS) // 64
#define TOK       (BATCH*SEQ)  // 2048 tokens
#define EPSF      1e-5f

typedef __bf16 bf16_t;
typedef __attribute__((ext_vector_type(16))) __bf16 v16bf;
typedef __attribute__((ext_vector_type(8)))  __bf16 v8bf;
typedef __attribute__((ext_vector_type(8)))  float  v8f;
typedef __attribute__((ext_vector_type(4)))  unsigned int v4u;
typedef __attribute__((ext_vector_type(8)))  int v8i;
typedef __attribute__((ext_vector_type(4)))  int v4i;

__device__ __forceinline__ float sigmoidf_(float x) { return 1.0f / (1.0f + __expf(-x)); }
__device__ __forceinline__ float siluf_(float x)    { return x * sigmoidf_(x); }
__device__ __forceinline__ float softplusf_(float x){ return (x > 20.0f) ? x : log1pf(__expf(x)); }

// ---------------------------------------------------------------------------
// Tensor Data Mover: issue one 2D tile load (64 rows x 32 bf16 cols) -> LDS.
// Descriptor per CDNA5 ISA ch.8 (D# group0/group1 bitfields); groups 2/3 and
// the trailing group are zero (2D tensor). Row stride = K elems, 2 B/elem.
// ---------------------------------------------------------------------------
__device__ __forceinline__ void tdm_issue_64x32(unsigned lds_byte_off,
                                                const bf16_t* gsrc, int K) {
    unsigned long long ga = (unsigned long long)(size_t)gsrc;
    v4u g0 = { 1u,                                  // count=1 (valid user desc)
               lds_byte_off,                        // lds_addr (bytes)
               (unsigned)(ga & 0xFFFFFFFFull),      // global_addr[31:0]
               (unsigned)(((ga >> 32) & 0x1FFFFFFull) | (2u << 30)) }; // addr[56:32], type=2
    v8i g1;
    g1[0] = (int)(1u << 16);                        // data_size=1 -> 2 bytes/elem
    g1[1] = (int)(((unsigned)K & 0xFFFFu) << 16);   // tensor_dim0[15:0]
    g1[2] = (int)((((unsigned)K >> 16) & 0xFFFFu) | (64u << 16)); // dim0 hi | tensor_dim1=64
    g1[3] = (int)(32u << 16);                       // tensor_dim1 hi=0 | tile_dim0=32
    g1[4] = 64;                                     // tile_dim1=64, tile_dim2=0
    g1[5] = K;                                      // tensor_dim0_stride[31:0] (elems)
    g1[6] = 0;                                      // stride hi / dim1_stride
    g1[7] = 0;
    v4i z4 = { 0, 0, 0, 0 };
    v8i z8 = { 0, 0, 0, 0, 0, 0, 0, 0 };
    __builtin_amdgcn_tensor_load_to_lds(g0, g1, z4, z4, z8, 0);
}

// ---------------------------------------------------------------------------
// Utility kernels
// ---------------------------------------------------------------------------
__global__ void k_zero(float* p, int n) {
    int i = blockIdx.x * blockDim.x + threadIdx.x;
    if (i < n) p[i] = 0.0f;
}

__global__ void k_cast_bf16(const float* __restrict__ src, bf16_t* __restrict__ dst, int n) {
    int i = blockIdx.x * blockDim.x + threadIdx.x;
    if (i < n) dst[i] = (bf16_t)src[i];
}

__global__ void k_negexp(const float* __restrict__ src, float* __restrict__ dst, int n) {
    int i = blockIdx.x * blockDim.x + threadIdx.x;
    if (i < n) dst[i] = -__expf(src[i]);
}

// ---------------------------------------------------------------------------
// LayerNorm (both inputs) + channel exchange + SE column-sum accumulation
// ---------------------------------------------------------------------------
__global__ __launch_bounds__(256)
void k_ln_exchange(const float* __restrict__ I1, const float* __restrict__ I2,
                   const float* __restrict__ w1, const float* __restrict__ b1,
                   const float* __restrict__ w2, const float* __restrict__ b2,
                   float* __restrict__ s1, float* __restrict__ s2,
                   float* __restrict__ colsum1, float* __restrict__ colsum2) {
    const int token = blockIdx.x;
    const int tid   = threadIdx.x;
    const int b     = token / SEQ;
    const size_t base = (size_t)token * DM;

    float x1a = I1[base + tid], x1b = I1[base + tid + 256];
    float x2a = I2[base + tid], x2b = I2[base + tid + 256];

    __shared__ float rs[4][256];
    rs[0][tid] = x1a + x1b;
    rs[1][tid] = x1a*x1a + x1b*x1b;
    rs[2][tid] = x2a + x2b;
    rs[3][tid] = x2a*x2a + x2b*x2b;
    __syncthreads();
    for (int off = 128; off > 0; off >>= 1) {
        if (tid < off) {
            rs[0][tid] += rs[0][tid+off];
            rs[1][tid] += rs[1][tid+off];
            rs[2][tid] += rs[2][tid+off];
            rs[3][tid] += rs[3][tid+off];
        }
        __syncthreads();
    }
    const float inv = 1.0f / (float)DM;
    float mu1 = rs[0][0] * inv, mu2 = rs[2][0] * inv;
    float v1  = rs[1][0] * inv - mu1*mu1;
    float v2  = rs[3][0] * inv - mu2*mu2;
    float r1  = rsqrtf(v1 + EPSF), r2 = rsqrtf(v2 + EPSF);

    #pragma unroll
    for (int half = 0; half < 2; ++half) {
        int c = tid + half*256;
        float xa = half ? x1b : x1a;
        float xb = half ? x2b : x2a;
        float n1 = (xa - mu1) * r1 * w1[c] + b1[c];
        float n2 = (xb - mu2) * r2 * w2[c] + b2[c];
        float o1 = ((c & 1) == 0) ? n2 : n1;
        float o2 = ((c & 1) == 0) ? n1 : n2;
        s1[base + c] = o1;
        s2[base + c] = o2;
        atomicAdd(&colsum1[b*DM + c], o1);
        atomicAdd(&colsum2[b*DM + c], o2);
    }
}

// ---------------------------------------------------------------------------
// SE gate: y = mean over tokens; gate = sigmoid(relu(y@w1^T)@w2^T)
// ---------------------------------------------------------------------------
__global__ __launch_bounds__(128)
void k_se_gate(const float* __restrict__ colsum,
               const float* __restrict__ w1,   // (128,512)
               const float* __restrict__ w2,   // (512,128)
               float* __restrict__ gate) {
    const int b   = blockIdx.x;
    const int tid = threadIdx.x;
    __shared__ float y[DM];
    __shared__ float hid[DM/4];
    const float invN = 1.0f / (float)SEQ;
    #pragma unroll
    for (int i = 0; i < 4; ++i) y[tid + i*128] = colsum[b*DM + tid + i*128] * invN;
    __syncthreads();
    float acc = 0.0f;
    const float* wr = w1 + (size_t)tid * DM;
    for (int c = 0; c < DM; ++c) acc += y[c] * wr[c];
    hid[tid] = fmaxf(acc, 0.0f);
    __syncthreads();
    #pragma unroll
    for (int i = 0; i < 4; ++i) {
        int c = tid + i*128;
        float a2 = 0.0f;
        const float* w2r = w2 + (size_t)c * (DM/4);
        for (int j = 0; j < DM/4; ++j) a2 += hid[j] * w2r[j];
        gate[b*DM + c] = sigmoidf_(a2);
    }
}

__global__ void k_scale_cast(const float* __restrict__ s, const float* __restrict__ gate,
                             bf16_t* __restrict__ xbf) {
    int i = blockIdx.x * blockDim.x + threadIdx.x;
    if (i >= TOK * DM) return;
    int row = i / DM, c = i % DM;
    int b = row / SEQ;
    xbf[i] = (bf16_t)(s[i] * gate[b*DM + c]);
}

// ---------------------------------------------------------------------------
// TDM + LDS double-buffered WMMA GEMM:  C[M,N] = A[M,K] @ W[N,K]^T (+epilogue)
//   block = 256 threads (8 waves) -> 64x64 macro-tile, K stepped by 32
//   wave w: tile row (w&3), tile cols {2*(w>>2), 2*(w>>2)+1}  (16x32 of C)
//   A/B K-slices staged into LDS by the Tensor Data Mover (wave 0 issues),
//   double buffered; s_wait_tensorcnt + barrier between stages.
//   A 32-bit LDS *offset* is laundered through inline asm each iteration:
//   the offset becomes opaque so the ds_load fragments must be re-issued per
//   K-step (the TDM's LDS writes are invisible to LLVM AA), while smem keeps
//   its LDS address space -> ds_load_b128, not flat loads.
//   mode 0: plain   mode 1: +bias[n], softplus   mode 2: +res[m,n]
// ---------------------------------------------------------------------------
__global__ __launch_bounds__(256)
void k_gemm_tdm(const bf16_t* __restrict__ A, const bf16_t* __restrict__ W,
                float* __restrict__ C, int M, int N, int K,
                int mode, const float* __restrict__ bias, const float* __restrict__ res) {
    // LDS: 2 buffers x (A 64x32 + B 64x32) bf16 = 16 KB
    __shared__ bf16_t smem[8192];
    const int bn  = blockIdx.x;     // N / 64
    const int bm  = blockIdx.y;     // M / 64
    const int tid = threadIdx.x;
    const int wid = tid >> 5;
    const int lane = tid & 31;
    const int ln = lane & 15;
    const int g  = lane >> 4;
    const int r  = wid & 3;         // tile row within macro-tile
    const int cp = wid >> 2;        // column pair selector
    const int c0 = cp * 2, c1 = cp * 2 + 1;

    // not provably dead (blockDim is runtime), never executes: gives the LDS
    // array a visible writer so loads from it are not undef-foldable.
    if (blockDim.x == 0) smem[tid] = (bf16_t)0.0f;

    const bf16_t* Abase = A + (size_t)(bm * 64) * K;
    const bf16_t* Wbase = W + (size_t)(bn * 64) * K;

    v8f acc0 = 0.0f, acc1 = 0.0f;

    // prologue: stage buffer 0 (K-step 0)
    if (tid < 32) {                       // wave 0 only (wave-uniform branch)
        tdm_issue_64x32(0u,    Abase, K);
        tdm_issue_64x32(4096u, Wbase, K);
    }
    __builtin_amdgcn_s_wait_tensorcnt(0);
    __syncthreads();

    int buf = 0;
    for (int k0 = 0; k0 < K; k0 += 32) {
        // stage next K-step into the other buffer while we compute
        if (k0 + 32 < K && tid < 32) {
            unsigned off = (unsigned)((buf ^ 1) * 8192);
            tdm_issue_64x32(off,        Abase + (k0 + 32), K);
            tdm_issue_64x32(off + 4096, Wbase + (k0 + 32), K);
        }

        // Launder the LDS element offset: opaque after the asm, so the
        // fragment loads below must be re-issued from LDS every iteration,
        // modeling the TDM's writes. smem keeps its LDS address space.
        unsigned sboff = (unsigned)(buf * 4096);
        asm volatile("" : "+v"(sboff) :: "memory");

        const bf16_t* sA = smem + sboff;               // elements
        const bf16_t* sB = smem + sboff + 2048;

        // A fragment: lane(m=ln,g): k = g*16 + e (contiguous along K)
        const int arow = r * 16 + ln;
        v16bf af = *(const v16bf*)(sA + arow * 32 + g * 16);

        // B fragments: lane(n=ln,g): k = g*16 + e (contiguous along K)
        v16bf bf0 = *(const v16bf*)(sB + (c0 * 16 + ln) * 32 + g * 16);
        v16bf bf1 = *(const v16bf*)(sB + (c1 * 16 + ln) * 32 + g * 16);

        acc0 = __builtin_amdgcn_wmma_f32_16x16x32_bf16(
                   false, af, false, bf0, (short)0, acc0, false, false);
        acc1 = __builtin_amdgcn_wmma_f32_16x16x32_bf16(
                   false, af, false, bf1, (short)0, acc1, false, false);

        __builtin_amdgcn_s_wait_tensorcnt(0);
        __syncthreads();
        buf ^= 1;
    }

    const int nG0 = bn * 64 + c0 * 16 + ln;
    const int nG1 = bn * 64 + c1 * 16 + ln;
    #pragma unroll
    for (int v = 0; v < 8; ++v) {
        int m = bm * 64 + r * 16 + v + 8 * g;
        float v0 = acc0[v], v1 = acc1[v];
        if (mode == 1) {
            v0 = softplusf_(v0 + bias[nG0]);
            v1 = softplusf_(v1 + bias[nG1]);
        } else if (mode == 2) {
            v0 += res[(size_t)m * N + nG0];
            v1 += res[(size_t)m * N + nG1];
        }
        C[(size_t)m * N + nG0] = v0;
        C[(size_t)m * N + nG1] = v1;
    }
}

// ---------------------------------------------------------------------------
// depthwise causal conv1d (DCONV=4) + bias + SiLU; writes f32 and bf16
// ---------------------------------------------------------------------------
__global__ void k_conv_silu(const float* __restrict__ xz,
                            const float* __restrict__ cw,
                            const float* __restrict__ cb,
                            float* __restrict__ xc, bf16_t* __restrict__ xcbf) {
    int i = blockIdx.x * blockDim.x + threadIdx.x;
    if (i >= TOK * DI) return;
    int row = i / DI, d = i % DI;
    int b = row / SEQ, t = row % SEQ;
    float acc = 0.0f;
    #pragma unroll
    for (int j = 0; j < DCONV; ++j) {
        int tt = t - (DCONV-1) + j;
        if (tt >= 0) acc += cw[d*DCONV + j] * xz[(size_t)(b*SEQ + tt) * (2*DI) + d];
    }
    float v = siluf_(acc + cb[d]);
    xc[i]   = v;
    xcbf[i] = (bf16_t)v;
}

__global__ void k_split_dt(const float* __restrict__ xdbl, bf16_t* __restrict__ dtbf) {
    int i = blockIdx.x * blockDim.x + threadIdx.x;
    if (i >= TOK * DTR) return;
    int row = i / DTR, r = i % DTR;
    dtbf[i] = (bf16_t)xdbl[(size_t)row * XDBL + r];
}

// ---------------------------------------------------------------------------
// selective scan: one lane per (b, d) channel; 16 states in VGPRs
// ---------------------------------------------------------------------------
__global__ __launch_bounds__(256)
void k_scan(const float* __restrict__ dtf,
            const float* __restrict__ xc,
            const float* __restrict__ xdbl,
            const float* __restrict__ Aexp,
            float* __restrict__ ys) {
    const int d = blockIdx.x * blockDim.x + threadIdx.x;
    const int b = blockIdx.y;
    float Arow[DS], h[DS];
    #pragma unroll
    for (int s = 0; s < DS; ++s) { Arow[s] = Aexp[d*DS + s]; h[s] = 0.0f; }
    for (int t = 0; t < SEQ; ++t) {
        const size_t row = (size_t)(b*SEQ + t);
        const float dtv = dtf[row*DI + d];
        const float xcv = xc [row*DI + d];
        const float* bc = xdbl + row*XDBL;
        float acc = 0.0f;
        #pragma unroll
        for (int s = 0; s < DS; ++s) {
            float dA  = __expf(dtv * Arow[s]);
            float dBu = dtv * bc[DTR + s] * xcv;
            h[s] = fmaf(dA, h[s], dBu);
            acc  = fmaf(h[s], bc[DTR + DS + s], acc);
        }
        ys[row*DI + d] = acc;
    }
}

__global__ void k_out_gate(const float* __restrict__ ys, const float* __restrict__ xc,
                           const float* __restrict__ xz, const float* __restrict__ Dw,
                           bf16_t* __restrict__ ybf) {
    int i = blockIdx.x * blockDim.x + threadIdx.x;
    if (i >= TOK * DI) return;
    int row = i / DI, d = i % DI;
    float z = xz[(size_t)row * (2*DI) + DI + d];
    float v = (ys[i] + xc[i] * Dw[d]) * siluf_(z);
    ybf[i] = (bf16_t)v;
}

// ---------------------------------------------------------------------------
// host orchestration
// ---------------------------------------------------------------------------
static inline int cdiv(int a, int b) { return (a + b - 1) / b; }

extern "C" void kernel_launch(void* const* d_in, const int* in_sizes, int n_in,
                              void* d_out, int out_size, void* d_ws, size_t ws_size,
                              hipStream_t stream) {
    const float* I1  = (const float*)d_in[0];
    const float* I2  = (const float*)d_in[1];
    const float* n1w = (const float*)d_in[2];
    const float* n1b = (const float*)d_in[3];
    const float* n2w = (const float*)d_in[4];
    const float* n2b = (const float*)d_in[5];
    const float* sew1= (const float*)d_in[6];
    const float* sew2= (const float*)d_in[7];
    float* out = (float*)d_out;

    char* p = (char*)d_ws;
    auto alloc = [&](size_t bytes) -> void* {
        void* r = (void*)p;
        p += (bytes + 255) & ~(size_t)255;
        return r;
    };
    float*  s1     = (float*) alloc((size_t)TOK*DM*4);
    float*  s2     = (float*) alloc((size_t)TOK*DM*4);
    float*  colsum = (float*) alloc((size_t)2*BATCH*DM*4);
    float*  colsum1= colsum;
    float*  colsum2= colsum + BATCH*DM;
    float*  gate1  = (float*) alloc((size_t)BATCH*DM*4);
    float*  gate2  = (float*) alloc((size_t)BATCH*DM*4);
    bf16_t* xbf1   = (bf16_t*)alloc((size_t)TOK*DM*2);
    bf16_t* xbf2   = (bf16_t*)alloc((size_t)TOK*DM*2);
    float*  xz     = (float*) alloc((size_t)TOK*2*DI*4);
    float*  xc     = (float*) alloc((size_t)TOK*DI*4);
    bf16_t* xcbf   = (bf16_t*)alloc((size_t)TOK*DI*2);
    float*  xdbl   = (float*) alloc((size_t)TOK*XDBL*4);
    bf16_t* dtbf   = (bf16_t*)alloc((size_t)TOK*DTR*2);
    float*  dtf    = (float*) alloc((size_t)TOK*DI*4);
    float*  ys     = (float*) alloc((size_t)TOK*DI*4);
    bf16_t* ybf    = (bf16_t*)alloc((size_t)TOK*DI*2);
    float*  Aexp   = (float*) alloc((size_t)DI*DS*4);
    bf16_t* w_in   = (bf16_t*)alloc((size_t)2*DI*DM*2);
    bf16_t* w_xp   = (bf16_t*)alloc((size_t)XDBL*DI*2);
    bf16_t* w_dt   = (bf16_t*)alloc((size_t)DI*DTR*2);
    bf16_t* w_out  = (bf16_t*)alloc((size_t)DM*DI*2);

    // ---- stage 1: LN + exchange + SE ----
    k_zero<<<cdiv(2*BATCH*DM,256),256,0,stream>>>(colsum, 2*BATCH*DM);
    k_ln_exchange<<<TOK,256,0,stream>>>(I1,I2,n1w,n1b,n2w,n2b,s1,s2,colsum1,colsum2);
    k_se_gate<<<BATCH,128,0,stream>>>(colsum1, sew1, sew2, gate1);
    k_se_gate<<<BATCH,128,0,stream>>>(colsum2, sew1, sew2, gate2);
    k_scale_cast<<<cdiv(TOK*DM,256),256,0,stream>>>(s1, gate1, xbf1);
    k_scale_cast<<<cdiv(TOK*DM,256),256,0,stream>>>(s2, gate2, xbf2);

    // ---- stage 2: two mamba mixers ----
    for (int mi = 0; mi < 2; ++mi) {
        const float* in_w   = (const float*)d_in[ 8 + 9*mi];
        const float* conv_w = (const float*)d_in[ 9 + 9*mi];
        const float* conv_b = (const float*)d_in[10 + 9*mi];
        const float* xproj_w= (const float*)d_in[11 + 9*mi];
        const float* dt_w   = (const float*)d_in[12 + 9*mi];
        const float* dt_b   = (const float*)d_in[13 + 9*mi];
        const float* A_log  = (const float*)d_in[14 + 9*mi];
        const float* Dw     = (const float*)d_in[15 + 9*mi];
        const float* out_w  = (const float*)d_in[16 + 9*mi];
        const float* resid  = (mi == 0) ? I1 : I2;
        bf16_t*      xbf    = (mi == 0) ? xbf1 : xbf2;
        float*       outp   = out + (size_t)mi * TOK * DM;

        k_cast_bf16<<<cdiv(2*DI*DM,256),256,0,stream>>>(in_w,   w_in,  2*DI*DM);
        k_cast_bf16<<<cdiv(XDBL*DI,256),256,0,stream>>>(xproj_w,w_xp,  XDBL*DI);
        k_cast_bf16<<<cdiv(DI*DTR,256),256,0,stream>>>(dt_w,   w_dt,  DI*DTR);
        k_cast_bf16<<<cdiv(DM*DI,256),256,0,stream>>>(out_w,  w_out, DM*DI);
        k_negexp   <<<cdiv(DI*DS,256),256,0,stream>>>(A_log,  Aexp,  DI*DS);

        // in_proj: xz[TOK,2048] = X @ in_w^T
        k_gemm_tdm<<<dim3((2*DI)/64, TOK/64),256,0,stream>>>(xbf, w_in, xz,
            TOK, 2*DI, DM, 0, nullptr, nullptr);
        // conv + silu
        k_conv_silu<<<cdiv(TOK*DI,256),256,0,stream>>>(xz, conv_w, conv_b, xc, xcbf);
        // x_proj: xdbl[TOK,64] = xc @ xproj_w^T
        k_gemm_tdm<<<dim3(XDBL/64, TOK/64),256,0,stream>>>(xcbf, w_xp, xdbl,
            TOK, XDBL, DI, 0, nullptr, nullptr);
        // dt slice + dt_proj (+bias, softplus)
        k_split_dt<<<cdiv(TOK*DTR,256),256,0,stream>>>(xdbl, dtbf);
        k_gemm_tdm<<<dim3(DI/64, TOK/64),256,0,stream>>>(dtbf, w_dt, dtf,
            TOK, DI, DTR, 1, dt_b, nullptr);
        // selective scan
        k_scan<<<dim3(DI/256, BATCH),256,0,stream>>>(dtf, xc, xdbl, Aexp, ys);
        // output gating
        k_out_gate<<<cdiv(TOK*DI,256),256,0,stream>>>(ys, xc, xz, Dw, ybf);
        // out_proj + residual -> d_out
        k_gemm_tdm<<<dim3(DM/64, TOK/64),256,0,stream>>>(ybf, w_out, outp,
            TOK, DM, DI, 2, nullptr, resid);
    }
}
